// LSTM_13812614824181
// MI455X (gfx1250) — compile-verified, matrix operand order checked by reading, not run on hardware
//
#include <hip/hip_runtime.h>
#include <hip/hip_bf16.h>
#include <math.h>

// ---------------------------------------------------------------------------
// LSTM on MI455X (gfx1250), wave32 + v_wmma_f32_16x16x32_bf16.
// Step t:  gate_g = [h_t | emb[x_t]] (256x1280 bf16) @ Wcat_g^T (1024x1280 bf16)
// fp32 accumulate; tanh/sigmoid/cell update fused in registers.
// Pipeline: global->regs (kc+1) || LDS->WMMA (kc, 1-deep B lookahead)
//           || regs->LDS (kc+1); XOR-toggled double-buffered LDS, unroll(1)
//           so the 8 accumulators keep a single register home (no mov storm).
// ---------------------------------------------------------------------------

#define B_  256
#define T_  128
#define D_  256
#define H_  1024
#define C_  10
#define K_  (H_ + D_)        // 1280 concat-K
#define KCH 32               // WMMA bf16 K per instruction
#define NKC (K_ / KCH)       // 40 K-chunks
#define ASTR 40              // LDS row stride (bf16): 80B = 20 banks, conflict-free
#define WSTR 40
#define ABUF (64 * ASTR)     // 2560 elements per A buffer
#define WBUF (4 * 64 * WSTR) // 10240 elements per W buffer

typedef __attribute__((ext_vector_type(16))) __bf16 v16bf;
typedef __attribute__((ext_vector_type(8)))  __bf16 v8bf;
typedef __attribute__((ext_vector_type(8)))  float  v8f;

// Gather one 16-lane fragment row slice per ISA 7.12.2 (bf16, K=32):
// lanes 0-15: K = 0..7 and 16..23 ; lanes 16-31: K = 8..15 and 24..31.
__device__ __forceinline__ v16bf load_frag(const __bf16* row, int kbIdx /*0|1*/) {
    const v8bf* p = reinterpret_cast<const v8bf*>(row);
    v8bf lo = p[kbIdx];
    v8bf hi = p[kbIdx + 2];
    return __builtin_shufflevector(lo, hi, 0,1,2,3,4,5,6,7,8,9,10,11,12,13,14,15);
}

__device__ __forceinline__ float sigmoidf_(float x) {
    return 1.0f / (1.0f + expf(-x));
}

#define WMMA_BF16(ACC, A, Bf) \
    ACC = __builtin_amdgcn_wmma_f32_16x16x32_bf16(false, (A), false, (Bf), \
                                                  (short)0, ACC, false, false)

// --------------------------- prep kernels ----------------------------------

// Wcat[g][o][k] bf16: k<1024 -> W*h[o][k], else W*x[o][k-1024].  g order {g,i,f,o}
__global__ void prep_wcat_kernel(const float* __restrict__ Wgh, const float* __restrict__ Wgx,
                                 const float* __restrict__ Wih, const float* __restrict__ Wix,
                                 const float* __restrict__ Wfh, const float* __restrict__ Wfx,
                                 const float* __restrict__ Woh, const float* __restrict__ Wox,
                                 __bf16* __restrict__ Wcat) {
    const float* Wh[4] = {Wgh, Wih, Wfh, Woh};
    const float* Wx[4] = {Wgx, Wix, Wfx, Wox};
    const int total = 4 * H_ * K_;
    for (int i = blockIdx.x * blockDim.x + threadIdx.x; i < total;
         i += gridDim.x * blockDim.x) {
        int g   = i / (H_ * K_);
        int rem = i - g * (H_ * K_);
        int o   = rem / K_;
        int k   = rem - o * K_;
        float v = (k < H_) ? Wh[g][o * H_ + k] : Wx[g][o * D_ + (k - H_)];
        Wcat[i] = (__bf16)v;
    }
}

// emb -> bf16, h0 -> 0 (bf16), c -> 0 (f32)
__global__ void prep_state_kernel(const float* __restrict__ emb,
                                  __bf16* __restrict__ embb,
                                  __bf16* __restrict__ h0,
                                  float*  __restrict__ c) {
    for (int i = blockIdx.x * blockDim.x + threadIdx.x; i < B_ * H_;
         i += gridDim.x * blockDim.x) {
        h0[i] = (__bf16)0.0f;
        c[i]  = 0.0f;
        if (i < 3 * D_) embb[i] = (__bf16)emb[i];
    }
}

// --------------------------- recurrent step --------------------------------
// grid: (16, 4) tiles of 64(H) x 64(B); block: 256 threads = 8 wave32s.
// wave w: rows [ (w&3)*16 , +16 ), cols [ (w>>2)*32 , +32 ), all 4 gates.
__global__ void __launch_bounds__(256)
lstm_step_kernel(const __bf16* __restrict__ Wcat,   // [4][H_][K_]
                 const __bf16* __restrict__ embb,   // [3][D_]
                 const int*    __restrict__ x,      // [B_][T_]
                 int t,
                 const float* __restrict__ bg, const float* __restrict__ bi,
                 const float* __restrict__ bfg, const float* __restrict__ bo,
                 const __bf16* __restrict__ h_in,   // [B_][H_]
                 __bf16*       __restrict__ h_out,  // [B_][H_]
                 float*        __restrict__ cbuf,   // [B_][H_]
                 float*        __restrict__ hfin) { // [B_][H_]
    __shared__ __align__(16) __bf16 A_lds[2 * ABUF];   // 10.0 KB (two buffers)
    __shared__ __align__(16) __bf16 W_lds[2 * WBUF];   // 40.0 KB (two buffers)

    const int tid    = threadIdx.x;
    const int lane   = tid & 31;
    const int wave   = tid >> 5;
    const int waveB  = wave & 3;              // 16-row slice
    const int waveN  = wave >> 2;             // 32-col slice
    const int lanelo = lane & 15;
    const int kbIdx  = (lane & 16) ? 1 : 0;
    const int bN     = blockIdx.x;            // H tile (0..15)
    const int bB     = blockIdx.y;            // B tile (0..3)

    // --- per-thread staging geometry (fixed for whole kernel) ---
    const int rA   = tid >> 2;                // A row 0..63
    const int segA = tid & 3;                 // 8-bf16 segment
    const int rowg = bB * 64 + rA;
    const int xidx = x[rowg * T_ + t];        // one gather per thread
    const __bf16* embrow = embb + xidx * D_;

    // 8 individually named accumulators: [gate g][ntile nt]
    v8f acc0_0 = {}, acc0_1 = {};             // gate g (tanh)
    v8f acc1_0 = {}, acc1_1 = {};             // gate i
    v8f acc2_0 = {}, acc2_1 = {};             // gate f
    v8f acc3_0 = {}, acc3_1 = {};             // gate o

    v8bf avreg;
    v8bf wvreg[4];

    // stage chunk kc's tiles from global into registers
    auto stage_to_regs = [&](int kc) {
        const int k = kc * KCH + segA * 8;
        avreg = (k < H_)
            ? *reinterpret_cast<const v8bf*>(h_in + (size_t)rowg * H_ + k)
            : *reinterpret_cast<const v8bf*>(embrow + (k - H_));
        #pragma unroll
        for (int it = 0; it < 4; ++it) {
            const int gid = tid + it * 256;
            const int g   = gid >> 8;
            const int r   = (gid >> 2) & 63;
            const int seg = gid & 3;
            const int kk  = kc * KCH + seg * 8;
            wvreg[it] = *reinterpret_cast<const v8bf*>(
                Wcat + ((size_t)g * H_ + bN * 64 + r) * K_ + kk);
        }
    };
    // commit staged registers into LDS at element offsets (offA, offW)
    auto regs_to_lds = [&](unsigned offA, unsigned offW) {
        *reinterpret_cast<v8bf*>(&A_lds[offA + rA * ASTR + segA * 8]) = avreg;
        #pragma unroll
        for (int it = 0; it < 4; ++it) {
            const int gid = tid + it * 256;
            const int g   = gid >> 8;
            const int r   = (gid >> 2) & 63;
            const int seg = gid & 3;
            *reinterpret_cast<v8bf*>(&W_lds[offW + (g * 64 + r) * WSTR + seg * 8]) =
                wvreg[it];
        }
    };

    stage_to_regs(0);
    regs_to_lds(0u, 0u);
    __syncthreads();

    unsigned aoff = 0, woff = 0;              // current compute buffer offsets
    #pragma unroll 1
    for (int kc = 0; kc < NKC; ++kc) {
        const bool more = (kc + 1) < NKC;
        if (more) stage_to_regs(kc + 1);      // global loads in flight

        // ---- compute chunk kc: 1-deep B-fragment lookahead, 8 WMMAs ----
        const __bf16* Ap = &A_lds[aoff];
        const __bf16* Wp = &W_lds[woff];
        const int cL0 = waveN * 32 + lanelo;  // nt=0 column ; nt=1 is +16
        const v16bf a = load_frag(Ap + (waveB * 16 + lanelo) * ASTR, kbIdx);

        v16bf bcur = load_frag(Wp + (0 * 64 + cL0) * WSTR, kbIdx);
        v16bf bnxt;
        bnxt = load_frag(Wp + (0 * 64 + cL0 + 16) * WSTR, kbIdx);
        WMMA_BF16(acc0_0, a, bcur);
        bcur = load_frag(Wp + (1 * 64 + cL0) * WSTR, kbIdx);
        WMMA_BF16(acc0_1, a, bnxt);
        bnxt = load_frag(Wp + (1 * 64 + cL0 + 16) * WSTR, kbIdx);
        WMMA_BF16(acc1_0, a, bcur);
        bcur = load_frag(Wp + (2 * 64 + cL0) * WSTR, kbIdx);
        WMMA_BF16(acc1_1, a, bnxt);
        bnxt = load_frag(Wp + (2 * 64 + cL0 + 16) * WSTR, kbIdx);
        WMMA_BF16(acc2_0, a, bcur);
        bcur = load_frag(Wp + (3 * 64 + cL0) * WSTR, kbIdx);
        WMMA_BF16(acc2_1, a, bnxt);
        bnxt = load_frag(Wp + (3 * 64 + cL0 + 16) * WSTR, kbIdx);
        WMMA_BF16(acc3_0, a, bcur);
        WMMA_BF16(acc3_1, a, bnxt);

        if (more) {
            regs_to_lds(aoff ^ ABUF, woff ^ WBUF);  // store into other buffer
            __syncthreads();                        // single barrier per iter
        }
        aoff ^= ABUF;                               // toggle buffers
        woff ^= WBUF;
    }

    // ---- fused gates + cell update (4 gate values are register-aligned) ----
    const int rbase = bB * 64 + waveB * 16 + ((lane & 16) ? 8 : 0);
    #pragma unroll
    for (int nt = 0; nt < 2; ++nt) {
        const v8f& ag = nt ? acc0_1 : acc0_0;
        const v8f& ai = nt ? acc1_1 : acc1_0;
        const v8f& af = nt ? acc2_1 : acc2_0;
        const v8f& ao = nt ? acc3_1 : acc3_0;
        const int col = bN * 64 + waveN * 32 + nt * 16 + lanelo;
        const float bgv = bg[col], biv = bi[col], bfv = bfg[col], bov = bo[col];
        #pragma unroll
        for (int e = 0; e < 8; ++e) {
            const int row = rbase + e;
            const size_t idx = (size_t)row * H_ + col;
            const float gv = tanhf(ag[e] + bgv);
            const float iv = sigmoidf_(ai[e] + biv);
            const float fv = sigmoidf_(af[e] + bfv);
            const float ov = sigmoidf_(ao[e] + bov);
            const float cn = gv * iv + cbuf[idx] * fv;
            const float hn = tanhf(cn) * ov;
            cbuf[idx]  = cn;
            hfin[idx]  = hn;
            h_out[idx] = (__bf16)hn;
        }
    }
}

// --------------------------- head + log_softmax ----------------------------
__global__ void head_kernel(const float* __restrict__ h,    // [B_][H_]
                            const float* __restrict__ Wph,  // [C_][H_]
                            const float* __restrict__ bp,   // [C_]
                            float* __restrict__ out) {      // [B_][C_]
    const int b   = blockIdx.x;
    const int tid = threadIdx.x;
    __shared__ float red[256];
    __shared__ float ps[C_];

    float partial[C_];
    #pragma unroll
    for (int j = 0; j < C_; ++j) partial[j] = 0.0f;
    for (int k = tid; k < H_; k += 256) {
        const float hv = h[(size_t)b * H_ + k];
        #pragma unroll
        for (int j = 0; j < C_; ++j) partial[j] += hv * Wph[j * H_ + k];
    }
    for (int j = 0; j < C_; ++j) {
        red[tid] = partial[j];
        __syncthreads();
        for (int s = 128; s > 0; s >>= 1) {
            if (tid < s) red[tid] += red[tid + s];
            __syncthreads();
        }
        if (tid == 0) ps[j] = red[0] + bp[j];
        __syncthreads();
    }
    if (tid == 0) {
        float m = ps[0];
        #pragma unroll
        for (int j = 1; j < C_; ++j) m = fmaxf(m, ps[j]);
        float s = 0.0f;
        #pragma unroll
        for (int j = 0; j < C_; ++j) s += expf(ps[j] - m);
        const float lse = m + logf(s);
        #pragma unroll
        for (int j = 0; j < C_; ++j) out[b * C_ + j] = ps[j] - lse;
    }
}

// ---------------------------------------------------------------------------
extern "C" void kernel_launch(void* const* d_in, const int* in_sizes, int n_in,
                              void* d_out, int out_size, void* d_ws, size_t ws_size,
                              hipStream_t stream) {
    const int*   x    = (const int*)  d_in[0];
    const float* emb  = (const float*)d_in[1];
    const float* Wgx  = (const float*)d_in[2];
    const float* Wgh  = (const float*)d_in[3];
    const float* bg   = (const float*)d_in[4];
    const float* Wix  = (const float*)d_in[5];
    const float* Wih  = (const float*)d_in[6];
    const float* bi   = (const float*)d_in[7];
    const float* Wfx  = (const float*)d_in[8];
    const float* Wfh  = (const float*)d_in[9];
    const float* bf_  = (const float*)d_in[10];
    const float* Wox  = (const float*)d_in[11];
    const float* Woh  = (const float*)d_in[12];
    const float* bo   = (const float*)d_in[13];
    const float* Wph  = (const float*)d_in[14];
    const float* bp   = (const float*)d_in[15];
    float* out = (float*)d_out;

    // Workspace layout (~13.6 MB, fully L2-resident on MI455X's 192 MB L2)
    __bf16* Wcat = (__bf16*)d_ws;                       // 4*1024*1280
    __bf16* embb = Wcat + (size_t)4 * H_ * K_;          // 3*256 (pad to 1024)
    __bf16* hb0  = embb + 1024;                         // 256*1024
    __bf16* hb1  = hb0 + (size_t)B_ * H_;               // 256*1024
    float*  cbuf = (float*)(hb1 + (size_t)B_ * H_);     // 256*1024 f32
    float*  hfin = cbuf + (size_t)B_ * H_;              // 256*1024 f32

    prep_wcat_kernel<<<512, 256, 0, stream>>>(Wgh, Wgx, Wih, Wix,
                                              Wfh, Wfx, Woh, Wox, Wcat);
    prep_state_kernel<<<256, 256, 0, stream>>>(emb, embb, hb0, cbuf);

    __bf16* hin  = hb0;
    __bf16* hout = hb1;
    for (int t = 0; t < T_; ++t) {
        lstm_step_kernel<<<dim3(16, 4), 256, 0, stream>>>(
            Wcat, embb, x, t, bg, bi, bf_, bo, hin, hout, cbuf, hfin);
        __bf16* tmp = hin; hin = hout; hout = tmp;
    }

    head_kernel<<<B_, 256, 0, stream>>>(hfin, Wph, bp, out);
}
